// MultiHeadedAttention_42640435315118
// MI455X (gfx1250) — compile-verified, matrix operand order checked by reading, not run on hardware
//
#include <hip/hip_runtime.h>
#include <hip/hip_bf16.h>

typedef __attribute__((ext_vector_type(16))) __bf16 v16bf;
typedef __attribute__((ext_vector_type(8)))  float  v8f;
typedef __attribute__((ext_vector_type(4)))  float  f32x4;
typedef __attribute__((ext_vector_type(4)))  unsigned int u32x4;
typedef __attribute__((ext_vector_type(2)))  unsigned int u32x2;

#define B_SZ  8
#define T_SEQ 1024
#define F_DIM 512
#define H_N   8
#define DK_N  64

union V16U { v16bf v; u32x4 u[2]; };
union BF4  { __bf16 b[4]; u32x2 u; };

// ---- WMMA helper: D = A(16x32 bf16) * B(32x16 bf16) + C(16x16 f32) ----
__device__ inline v8f wmma_bf16(v16bf a, v16bf b, v8f c) {
  return __builtin_amdgcn_wmma_f32_16x16x32_bf16(false, a, false, b,
                                                 (short)0, c, false, false);
}

// Fragment from row-major bf16 (row stride ld, 16B-aligned rows). CDNA5 16-bit
// layout: lane&15 = M/N index, lane>>4 selects K-halves {0..7,16..23} vs
// {8..15,24..31}. Two global_load_b128 per fragment.
__device__ inline v16bf ld_frag_g(const __bf16* __restrict__ base, int ld,
                                  int lane, int kkbase) {
  int r  = lane & 15;
  int k0 = kkbase + ((lane >> 4) << 3);
  const __bf16* p = base + (size_t)r * ld + k0;
  V16U t;
  t.u[0] = *(const u32x4*)(p);
  t.u[1] = *(const u32x4*)(p + 16);
  return t.v;
}

// Same pattern from LDS; ld must be a multiple of 8 elements -> ds_load_b128.
__device__ inline v16bf lds_frag(const __bf16* base, int ld, int lane) {
  int r  = lane & 15;
  int kh = (lane >> 4) << 3;
  const __bf16* p = base + r * ld + kh;
  V16U t;
  t.u[0] = *(const u32x4*)(p);
  t.u[1] = *(const u32x4*)(p + 16);
  return t.v;
}

// ---------------- Projection GEMM: C = X * W^T + b, bf16 out ---------------
// VT==0: head-major [B,H,T,DK]; VT==1: transposed [B,H,DK,T].
template <int VT>
__global__ __launch_bounds__(256)
void proj_gemm_kernel(const float* __restrict__ X,   // [8192,512]
                      const float* __restrict__ W,   // [512,512], uses W[n][k]
                      const float* __restrict__ bias,
                      __bf16* __restrict__ Out) {
  __shared__ __bf16 As[128][40];
  __shared__ __bf16 Bs[64][40];
  const int m0 = blockIdx.x * 128;
  const int n0 = blockIdx.y * 64;
  const int wave = threadIdx.x >> 5, lane = threadIdx.x & 31;
  const int tid = threadIdx.x;
  v8f acc[4] = {};

  for (int k0 = 0; k0 < F_DIM; k0 += 32) {
#pragma unroll
    for (int i = 0; i < 4; ++i) {            // A tile 128x32: f32x4 -> 4xbf16
      int idx4 = tid + i * 256;              // 1024 chunks of 4
      int r = idx4 >> 3, c4 = (idx4 & 7) << 2;
      f32x4 x = *(const f32x4*)(X + (size_t)(m0 + r) * F_DIM + k0 + c4);
      BF4 t;
      t.b[0] = (__bf16)x[0]; t.b[1] = (__bf16)x[1];
      t.b[2] = (__bf16)x[2]; t.b[3] = (__bf16)x[3];
      *(u32x2*)&As[r][c4] = t.u;
    }
#pragma unroll
    for (int i = 0; i < 2; ++i) {            // B tile 64x32 from W[n][k]
      int idx4 = tid + i * 256;              // 512 chunks of 4
      int r = idx4 >> 3, c4 = (idx4 & 7) << 2;
      f32x4 x = *(const f32x4*)(W + (size_t)(n0 + r) * F_DIM + k0 + c4);
      BF4 t;
      t.b[0] = (__bf16)x[0]; t.b[1] = (__bf16)x[1];
      t.b[2] = (__bf16)x[2]; t.b[3] = (__bf16)x[3];
      *(u32x2*)&Bs[r][c4] = t.u;
    }
    __syncthreads();
    v16bf af = lds_frag(&As[wave * 16][0], 40, lane);
#pragma unroll
    for (int d = 0; d < 4; ++d) {
      v16bf bf = lds_frag(&Bs[d * 16][0], 40, lane);
      acc[d] = wmma_bf16(af, bf, acc[d]);
    }
    __syncthreads();
  }
  const int rbase = (lane >> 4) << 3, colj = lane & 15;
#pragma unroll
  for (int d = 0; d < 4; ++d) {
    int n = n0 + d * 16 + colj;
    float bv = bias[n];
    int h = n >> 6, dk = n & 63;
#pragma unroll
    for (int g = 0; g < 8; ++g) {
      int m = m0 + wave * 16 + rbase + g;
      int bb = m >> 10, t = m & 1023;
      size_t off;
      if (VT)
        off = ((((size_t)bb * H_N + h) * DK_N) + dk) * T_SEQ + t;
      else
        off = ((((size_t)bb * H_N + h) * T_SEQ) + t) * DK_N + dk;
      Out[off] = (__bf16)(acc[d][g] + bv);
    }
  }
}

// ---------------- Output GEMM: out = ctx(bf16) * Wo^T + bo (f32) ----------
__global__ __launch_bounds__(256)
void out_gemm_kernel(const __bf16* __restrict__ Xb,  // [8192,512] bf16
                     const float* __restrict__ W,    // [512,512]
                     const float* __restrict__ bias,
                     float* __restrict__ Out) {      // [8192,512] f32
  __shared__ __bf16 As[128][40];
  __shared__ __bf16 Bs[64][40];
  const int m0 = blockIdx.x * 128;
  const int n0 = blockIdx.y * 64;
  const int wave = threadIdx.x >> 5, lane = threadIdx.x & 31;
  const int tid = threadIdx.x;
  v8f acc[4] = {};

  for (int k0 = 0; k0 < F_DIM; k0 += 32) {
#pragma unroll
    for (int i = 0; i < 2; ++i) {            // A tile 128x32 bf16: b128 copies
      int idx8 = tid + i * 256;              // 512 chunks of 8
      int r = idx8 >> 2, c8 = (idx8 & 3) << 3;
      *(u32x4*)&As[r][c8] =
          *(const u32x4*)(Xb + (size_t)(m0 + r) * F_DIM + k0 + c8);
    }
#pragma unroll
    for (int i = 0; i < 2; ++i) {            // B tile 64x32 from Wo
      int idx4 = tid + i * 256;
      int r = idx4 >> 3, c4 = (idx4 & 7) << 2;
      f32x4 x = *(const f32x4*)(W + (size_t)(n0 + r) * F_DIM + k0 + c4);
      BF4 t;
      t.b[0] = (__bf16)x[0]; t.b[1] = (__bf16)x[1];
      t.b[2] = (__bf16)x[2]; t.b[3] = (__bf16)x[3];
      *(u32x2*)&Bs[r][c4] = t.u;
    }
    __syncthreads();
    v16bf af = lds_frag(&As[wave * 16][0], 40, lane);
#pragma unroll
    for (int d = 0; d < 4; ++d) {
      v16bf bf = lds_frag(&Bs[d * 16][0], 40, lane);
      acc[d] = wmma_bf16(af, bf, acc[d]);
    }
    __syncthreads();
  }
  const int rbase = (lane >> 4) << 3, colj = lane & 15;
#pragma unroll
  for (int d = 0; d < 4; ++d) {
    int n = n0 + d * 16 + colj;
    float bv = bias[n];
#pragma unroll
    for (int g = 0; g < 8; ++g) {
      int m = m0 + wave * 16 + rbase + g;
      Out[(size_t)m * F_DIM + n] = acc[d][g] + bv;
    }
  }
}

// ---------------- softmax_plus scale per batch ----------------------------
__global__ void mask_count_kernel(const unsigned char* __restrict__ mask,
                                  const float* __restrict__ tll,
                                  float* __restrict__ scb) {
  __shared__ int red[256];
  int b = blockIdx.x, t = threadIdx.x;
  int s = 0;
  for (int i = t; i < T_SEQ; i += 256) s += (mask[b * T_SEQ + i] != 0);
  red[t] = s; __syncthreads();
  for (int k = 128; k > 0; k >>= 1) {
    if (t < k) red[t] += red[t + k];
    __syncthreads();
  }
  if (t == 0) {
    float l = fmaxf((float)red[0], 1.0f);
    scb[b] = __logf(l) / tll[0];
  }
}

// ---------------- bias helpers --------------------------------------------
__device__ inline float t5_bias(int tq, int tk, const float* __restrict__ re) {
  int n = tq - tk;                       // n = -rel_pos
  int ret = (n < 0) ? 16 : 0;
  int an = (n < 0) ? -n : n;
  int bucket;
  if (an < 8) bucket = ret + an;
  else {
    int vl = 8 + (int)(__logf((float)an * 0.125f) * (8.0f / 2.7725887f));
    bucket = ret + (vl < 15 ? vl : 15);
  }
  return re[bucket] * 8.0f;              // * sqrt(DK)
}

__device__ inline float dis_bias(int tq, int tk, float omiga, float gb) {
  float d = (float)(tk - tq);
  return -fabsf(fabsf(d * d * omiga) - fabsf(gb));
}

// ---------------- Flash attention (per (b,h), 128 q-rows / block) ---------
__global__ __launch_bounds__(256)
void flash_attn_kernel(const __bf16* __restrict__ Q,   // [B,H,T,DK]
                       const __bf16* __restrict__ Kt,  // [B,H,T,DK]
                       const __bf16* __restrict__ VT,  // [B,H,DK,T] transposed
                       const float* __restrict__ rel_emb,
                       const unsigned char* __restrict__ mask, // [B,1,T]
                       const float* __restrict__ omiga_p,
                       const float* __restrict__ gbias_p,
                       const float* __restrict__ scb,  // [B]
                       __bf16* __restrict__ ctx) {     // [B,T,F]
  const int bh = blockIdx.y;
  const int b = bh >> 3, h = bh & 7;
  const int wave = threadIdx.x >> 5, lane = threadIdx.x & 31;
  const int q0 = blockIdx.x * 128 + wave * 16;

  const __bf16* Qbh = Q  + (size_t)bh * T_SEQ * DK_N;
  const __bf16* Kbh = Kt + (size_t)bh * T_SEQ * DK_N;
  const __bf16* Vbh = VT + (size_t)bh * T_SEQ * DK_N;  // [DK][T]
  const unsigned char* mb = mask + (size_t)b * T_SEQ;

  const float omiga = omiga_p[0], gb = gbias_p[0], sc = scb[b];

  __shared__ __bf16 Pbuf[8][16][32];     // per-wave probability tile

  v16bf qf0 = ld_frag_g(Qbh + (size_t)q0 * DK_N, DK_N, lane, 0);
  v16bf qf1 = ld_frag_g(Qbh + (size_t)q0 * DK_N, DK_N, lane, 32);

  v8f acc[4] = {};
  float mrow[8], lrow[8];
#pragma unroll
  for (int g = 0; g < 8; ++g) { mrow[g] = -3.0e38f; lrow[g] = 0.0f; }

  const int rbase = (lane >> 4) << 3, colj = lane & 15;

  for (int kt = 0; kt < T_SEQ; kt += 32) {
    // ---- S = Q * K^T for 16x32 key tile ----
    v8f s0 = {}, s1 = {};
    {
      v16bf b00 = ld_frag_g(Kbh + (size_t)kt * DK_N,        DK_N, lane, 0);
      v16bf b10 = ld_frag_g(Kbh + (size_t)(kt + 16) * DK_N, DK_N, lane, 0);
      s0 = wmma_bf16(qf0, b00, s0);
      s1 = wmma_bf16(qf0, b10, s1);
      v16bf b01 = ld_frag_g(Kbh + (size_t)kt * DK_N,        DK_N, lane, 32);
      v16bf b11 = ld_frag_g(Kbh + (size_t)(kt + 16) * DK_N, DK_N, lane, 32);
      s0 = wmma_bf16(qf1, b01, s0);
      s1 = wmma_bf16(qf1, b11, s1);
    }
    const int tkA = kt + colj, tkB = kt + 16 + colj;
    const bool vA = mb[tkA] != 0, vB = mb[tkB] != 0;

    // ---- biases + softmax_plus scaling + online softmax ----
#pragma unroll
    for (int g = 0; g < 8; ++g) {
      int tq = q0 + rbase + g;
      float sA = (s0[g] + t5_bias(tq, tkA, rel_emb) + dis_bias(tq, tkA, omiga, gb)) * 0.125f;
      float sB = (s1[g] + t5_bias(tq, tkB, rel_emb) + dis_bias(tq, tkB, omiga, gb)) * 0.125f;
      sA = vA ? sA * sc : -1.0e30f;
      sB = vB ? sB * sc : -1.0e30f;

      float tmax = fmaxf(sA, sB);
#pragma unroll
      for (int off = 1; off <= 8; off <<= 1)
        tmax = fmaxf(tmax, __shfl_xor(tmax, off));     // 16-lane group reduce

      float mn = fmaxf(mrow[g], tmax);
      float rescale = __expf(mrow[g] - mn);
      float pA = __expf(sA - mn), pB = __expf(sB - mn);
      float psum = pA + pB;
#pragma unroll
      for (int off = 1; off <= 8; off <<= 1)
        psum += __shfl_xor(psum, off);

      lrow[g] = lrow[g] * rescale + psum;
      mrow[g] = mn;
#pragma unroll
      for (int d = 0; d < 4; ++d) acc[d][g] = acc[d][g] * rescale;

      Pbuf[wave][rbase + g][colj]      = (__bf16)pA;
      Pbuf[wave][rbase + g][16 + colj] = (__bf16)pB;
    }
    __syncthreads();   // order P stores before fragment reads (uniform)

    // ---- acc += P * V : V^T rows are dk, contiguous over keys ----
    v16bf pf = lds_frag(&Pbuf[wave][0][0], 32, lane);
#pragma unroll
    for (int d = 0; d < 4; ++d) {
      v16bf vf = ld_frag_g(Vbh + (size_t)(d * 16) * T_SEQ + kt, T_SEQ, lane, 0);
      acc[d] = wmma_bf16(pf, vf, acc[d]);
    }
  }

  // ---- normalize and write context [B,T,H*DK] ----
#pragma unroll
  for (int g = 0; g < 8; ++g) {
    float inv = lrow[g] > 0.0f ? 1.0f / lrow[g] : 0.0f;
    int tq = q0 + rbase + g;
#pragma unroll
    for (int d = 0; d < 4; ++d) {
      int col = h * DK_N + d * 16 + colj;
      ctx[((size_t)b * T_SEQ + tq) * F_DIM + col] = (__bf16)(acc[d][g] * inv);
    }
  }
}

// ---------------- host-side launcher --------------------------------------
extern "C" void kernel_launch(void* const* d_in, const int* in_sizes, int n_in,
                              void* d_out, int out_size, void* d_ws, size_t ws_size,
                              hipStream_t stream) {
  const float* query = (const float*)d_in[0];
  const float* key   = (const float*)d_in[1];
  const float* value = (const float*)d_in[2];
  const unsigned char* mask = (const unsigned char*)d_in[3];
  const float* Wq = (const float*)d_in[4];
  const float* bq = (const float*)d_in[5];
  const float* Wk = (const float*)d_in[6];
  const float* bk = (const float*)d_in[7];
  const float* Wv = (const float*)d_in[8];
  const float* bv = (const float*)d_in[9];
  const float* Wo = (const float*)d_in[10];
  const float* bo = (const float*)d_in[11];
  const float* rel_emb = (const float*)d_in[12];
  const float* omiga   = (const float*)d_in[13];
  const float* g_bias  = (const float*)d_in[14];
  const float* tll     = (const float*)d_in[15];

  char* w = (char*)d_ws;
  const size_t MB8 = (size_t)8 << 20;            // 4M bf16 elems
  __bf16* Qh  = (__bf16*)(w);
  __bf16* Kh  = (__bf16*)(w + MB8);
  __bf16* VTh = (__bf16*)(w + 2 * MB8);
  __bf16* Ctx = (__bf16*)(w + 3 * MB8);
  float*  scb = (float*)(w + 4 * MB8);

  dim3 gg(64, 8);                                // 8192/128 x 512/64
  proj_gemm_kernel<0><<<gg, 256, 0, stream>>>(query, Wq, bq, Qh);
  proj_gemm_kernel<0><<<gg, 256, 0, stream>>>(key,   Wk, bk, Kh);
  proj_gemm_kernel<1><<<gg, 256, 0, stream>>>(value, Wv, bv, VTh);

  mask_count_kernel<<<B_SZ, 256, 0, stream>>>(mask, tll, scb);

  flash_attn_kernel<<<dim3(8, 64), 256, 0, stream>>>(Qh, Kh, VTh, rel_emb, mask,
                                                     omiga, g_bias, scb, Ctx);

  out_gemm_kernel<<<gg, 256, 0, stream>>>(Ctx, Wo, bo, (float*)d_out);
}
